// DSFusion_84902913507853
// MI455X (gfx1250) — matrix operand order repaired and checked by reading, not compiled, not run on hardware
//
#include <hip/hip_runtime.h>
#include <hip/hip_bf16.h>

#define INPUT_DIM 2048
#define CLASSES   1000
#define BATCH     8192

typedef __attribute__((ext_vector_type(16))) __bf16 v16bf;
typedef __attribute__((ext_vector_type(8)))  __bf16 v8bf;
typedef __attribute__((ext_vector_type(8)))  float  v8f;
typedef __attribute__((ext_vector_type(4)))  float  f32x4;

// ===========================================================================
// fp32 -> bf16 streaming conversion (8 elements / thread, b128 in, b128 out)
// ===========================================================================
__global__ __launch_bounds__(256)
void cvt_f32_bf16_kernel(const float* __restrict__ src,
                         __bf16* __restrict__ dst, int n8) {
  int i = (blockIdx.x * 256 + threadIdx.x);
  if (i < n8) {
    const f32x4* p = (const f32x4*)(src + (size_t)i * 8);
    f32x4 a = p[0], b = p[1];
    v8bf o;
    o[0] = (__bf16)a[0]; o[1] = (__bf16)a[1];
    o[2] = (__bf16)a[2]; o[3] = (__bf16)a[3];
    o[4] = (__bf16)b[0]; o[5] = (__bf16)b[1];
    o[6] = (__bf16)b[2]; o[7] = (__bf16)b[3];
    *(v8bf*)(dst + (size_t)i * 8) = o;
  }
}

// ===========================================================================
// bf16 tile loaders (pre-converted operands) — pure b128 loads, zero VALU.
// A-tile (16x32 MxK), contiguous K order, same as B:
//   lane L:    M=L, elems 0-15 = K 0..15
//   lane L+16: M=L, elems 0-15 = K 16..31
// B-tile (32x16 KxN), contiguous K order, same as A:
//   lanes 0-15: N=L, 16 contiguous K (0..15); lanes 16-31: K 16..31
// ===========================================================================
__device__ __forceinline__ v16bf load_a_tile_bf(const __bf16* __restrict__ row,
                                                int k0, int hi) {
  v8bf lo = *(const v8bf*)(row + k0 + hi * 16);
  v8bf hh = *(const v8bf*)(row + k0 + hi * 16 + 8);
  return __builtin_shufflevector(lo, hh, 0, 1, 2, 3, 4, 5, 6, 7,
                                         8, 9, 10, 11, 12, 13, 14, 15);
}

__device__ __forceinline__ v16bf load_b_tile_bf(const __bf16* __restrict__ row,
                                                int k0, int hi) {
  return *(const v16bf*)(row + k0 + hi * 16);
}

// fp32 tile loaders (fallback path: convert on the fly)
__device__ __forceinline__ v16bf load_a_tile_f32(const float* __restrict__ row,
                                                 int k0, int hi) {
  const f32x4* p0 = (const f32x4*)(row + k0 + hi * 16);
  const f32x4* p1 = (const f32x4*)(row + k0 + hi * 16 + 8);
  f32x4 c0 = p0[0], c1 = p0[1], c2 = p1[0], c3 = p1[1];
  v16bf t;
  t[0]  = (__bf16)c0[0]; t[1]  = (__bf16)c0[1]; t[2]  = (__bf16)c0[2]; t[3]  = (__bf16)c0[3];
  t[4]  = (__bf16)c1[0]; t[5]  = (__bf16)c1[1]; t[6]  = (__bf16)c1[2]; t[7]  = (__bf16)c1[3];
  t[8]  = (__bf16)c2[0]; t[9]  = (__bf16)c2[1]; t[10] = (__bf16)c2[2]; t[11] = (__bf16)c2[3];
  t[12] = (__bf16)c3[0]; t[13] = (__bf16)c3[1]; t[14] = (__bf16)c3[2]; t[15] = (__bf16)c3[3];
  return t;
}

__device__ __forceinline__ v16bf load_b_tile_f32(const float* __restrict__ row,
                                                 int k0, int hi) {
  const f32x4* p = (const f32x4*)(row + k0 + hi * 16);
  f32x4 c0 = p[0], c1 = p[1], c2 = p[2], c3 = p[3];
  v16bf t;
  t[0]  = (__bf16)c0[0]; t[1]  = (__bf16)c0[1]; t[2]  = (__bf16)c0[2]; t[3]  = (__bf16)c0[3];
  t[4]  = (__bf16)c1[0]; t[5]  = (__bf16)c1[1]; t[6]  = (__bf16)c1[2]; t[7]  = (__bf16)c1[3];
  t[8]  = (__bf16)c2[0]; t[9]  = (__bf16)c2[1]; t[10] = (__bf16)c2[2]; t[11] = (__bf16)c2[3];
  t[12] = (__bf16)c3[0]; t[13] = (__bf16)c3[1]; t[14] = (__bf16)c3[2]; t[15] = (__bf16)c3[3];
  return t;
}

// Shared epilogue: bias + stable softplus + 1, masked store.
__device__ __forceinline__ void store_alpha(v8f accs[2][2],
                                            const float* __restrict__ bv,
                                            float* __restrict__ o,
                                            int m0, int n0, int hi, int r16) {
#pragma unroll
  for (int t = 0; t < 2; ++t) {
    const int mBase = m0 + t * 16 + hi * 8;
#pragma unroll
    for (int u = 0; u < 2; ++u) {
      const int n = n0 + u * 16 + r16;
      if (n < CLASSES) {
        const float bias = bv[n];
#pragma unroll
        for (int r = 0; r < 8; ++r) {
          float v  = accs[t][u][r] + bias;
          float sp = fmaxf(v, 0.0f) + __logf(1.0f + __expf(-fabsf(v)));
          o[(size_t)(mBase + r) * CLASSES + n] = sp + 1.0f;
        }
      }
    }
  }
}

// ===========================================================================
// Kernel 1 (primary): bf16-operand GEMM + bias + softplus + 1.
// Block = 256 thr = 8 waves as 4(M) x 2(N); block tile 128M x 64N.
// Wave = 32x32 tile = 2x2 v_wmma_f32_16x16x32_bf16 accumulators.
// Inner loop per K-step of 32: 8 x b128 loads, 4 WMMAs, no conversions.
// ===========================================================================
__global__ __launch_bounds__(256)
void gemm_softplus_bf16_kernel(const __bf16* __restrict__ xb,
                               const __bf16* __restrict__ yb,
                               const __bf16* __restrict__ Wxb,
                               const float*  __restrict__ bx,
                               const __bf16* __restrict__ Wyb,
                               const float*  __restrict__ by,
                               float* __restrict__ alpha_x,
                               float* __restrict__ alpha_y) {
  const int lane = threadIdx.x & 31;
  const int wid  = threadIdx.x >> 5;
  const int hi   = lane >> 4;
  const int r16  = lane & 15;
  const int wm   = wid & 3;
  const int wn   = wid >> 2;

  const int m0 = blockIdx.y * 128 + wm * 32;
  const int n0 = blockIdx.x * 64  + wn * 32;

  const __bf16* __restrict__ A  = (blockIdx.z == 0) ? xb  : yb;
  const __bf16* __restrict__ W  = (blockIdx.z == 0) ? Wxb : Wyb;
  const float*  __restrict__ bv = (blockIdx.z == 0) ? bx  : by;
  float*        __restrict__ o  = (blockIdx.z == 0) ? alpha_x : alpha_y;

  const __bf16* aRow0 = A + (size_t)(m0 + r16)      * INPUT_DIM;
  const __bf16* aRow1 = A + (size_t)(m0 + 16 + r16) * INPUT_DIM;

  const int n_t0 = n0 + r16;
  const int n_t1 = n0 + 16 + r16;
  const int nc0  = (n_t0 < CLASSES) ? n_t0 : (CLASSES - 1);
  const int nc1  = (n_t1 < CLASSES) ? n_t1 : (CLASSES - 1);
  const __bf16* bRow0 = W + (size_t)nc0 * INPUT_DIM;
  const __bf16* bRow1 = W + (size_t)nc1 * INPUT_DIM;

  v8f acc00 = {}, acc01 = {}, acc10 = {}, acc11 = {};

  for (int k0 = 0; k0 < INPUT_DIM; k0 += 32) {
    if (k0 + 64 < INPUT_DIM) {         // pull next K-chunks toward near caches
      __builtin_prefetch(aRow0 + k0 + 64, 0, 3);
      __builtin_prefetch(aRow1 + k0 + 64, 0, 3);
    }
    v16bf a0 = load_a_tile_bf(aRow0, k0, hi);
    v16bf a1 = load_a_tile_bf(aRow1, k0, hi);
    v16bf b0 = load_b_tile_bf(bRow0, k0, hi);
    v16bf b1 = load_b_tile_bf(bRow1, k0, hi);

    acc00 = __builtin_amdgcn_wmma_f32_16x16x32_bf16(false, a0, false, b0,
                                                    (short)0, acc00, false, false);
    acc01 = __builtin_amdgcn_wmma_f32_16x16x32_bf16(false, a0, false, b1,
                                                    (short)0, acc01, false, false);
    acc10 = __builtin_amdgcn_wmma_f32_16x16x32_bf16(false, a1, false, b0,
                                                    (short)0, acc10, false, false);
    acc11 = __builtin_amdgcn_wmma_f32_16x16x32_bf16(false, a1, false, b1,
                                                    (short)0, acc11, false, false);
  }

  v8f accs[2][2] = {{acc00, acc01}, {acc10, acc11}};
  store_alpha(accs, bv, o, m0, n0, hi, r16);
}

// ===========================================================================
// Kernel 1 (fallback, used only if workspace too small): fp32 source,
// convert-on-the-fly. Identical math.
// ===========================================================================
__global__ __launch_bounds__(256)
void gemm_softplus_fused_kernel(const float* __restrict__ x,
                                const float* __restrict__ y,
                                const float* __restrict__ Wx,
                                const float* __restrict__ bx,
                                const float* __restrict__ Wy,
                                const float* __restrict__ by,
                                float* __restrict__ alpha_x,
                                float* __restrict__ alpha_y) {
  const int lane = threadIdx.x & 31;
  const int wid  = threadIdx.x >> 5;
  const int hi   = lane >> 4;
  const int r16  = lane & 15;
  const int wm   = wid & 3;
  const int wn   = wid >> 2;

  const int m0 = blockIdx.y * 128 + wm * 32;
  const int n0 = blockIdx.x * 64  + wn * 32;

  const float* __restrict__ A  = (blockIdx.z == 0) ? x  : y;
  const float* __restrict__ W  = (blockIdx.z == 0) ? Wx : Wy;
  const float* __restrict__ bv = (blockIdx.z == 0) ? bx : by;
  float*       __restrict__ o  = (blockIdx.z == 0) ? alpha_x : alpha_y;

  const float* aRow0 = A + (size_t)(m0 + r16)      * INPUT_DIM;
  const float* aRow1 = A + (size_t)(m0 + 16 + r16) * INPUT_DIM;

  const int n_t0 = n0 + r16;
  const int n_t1 = n0 + 16 + r16;
  const int nc0  = (n_t0 < CLASSES) ? n_t0 : (CLASSES - 1);
  const int nc1  = (n_t1 < CLASSES) ? n_t1 : (CLASSES - 1);
  const float* bRow0 = W + (size_t)nc0 * INPUT_DIM;
  const float* bRow1 = W + (size_t)nc1 * INPUT_DIM;

  v8f acc00 = {}, acc01 = {}, acc10 = {}, acc11 = {};

  for (int k0 = 0; k0 < INPUT_DIM; k0 += 32) {
    if (k0 + 32 < INPUT_DIM) {
      __builtin_prefetch(aRow0 + k0 + 32, 0, 3);
      __builtin_prefetch(aRow1 + k0 + 32, 0, 3);
    }
    v16bf a0 = load_a_tile_f32(aRow0, k0, hi);
    v16bf a1 = load_a_tile_f32(aRow1, k0, hi);
    v16bf b0 = load_b_tile_f32(bRow0, k0, hi);
    v16bf b1 = load_b_tile_f32(bRow1, k0, hi);

    acc00 = __builtin_amdgcn_wmma_f32_16x16x32_bf16(false, a0, false, b0,
                                                    (short)0, acc00, false, false);
    acc01 = __builtin_amdgcn_wmma_f32_16x16x32_bf16(false, a0, false, b1,
                                                    (short)0, acc01, false, false);
    acc10 = __builtin_amdgcn_wmma_f32_16x16x32_bf16(false, a1, false, b0,
                                                    (short)0, acc10, false, false);
    acc11 = __builtin_amdgcn_wmma_f32_16x16x32_bf16(false, a1, false, b1,
                                                    (short)0, acc11, false, false);
  }

  v8f accs[2][2] = {{acc00, acc01}, {acc10, acc11}};
  store_alpha(accs, bv, o, m0, n0, hi, r16);
}

// ===========================================================================
// Kernel 2: Dempster-Shafer combination, one block (256 thr) per batch row.
// Single fused reduction (S1, S2, sum((a1-1)(a2-1))); closed forms for sum(b).
// ===========================================================================
__global__ __launch_bounds__(256)
void ds_combine_kernel(const float* __restrict__ ax,
                       const float* __restrict__ ay,
                       float* __restrict__ aa) {
  const int row = blockIdx.x;
  const float* __restrict__ a1 = ax + (size_t)row * CLASSES;
  const float* __restrict__ a2 = ay + (size_t)row * CLASSES;

  float s1 = 0.0f, s2 = 0.0f, pp = 0.0f;
  for (int c = threadIdx.x; c < CLASSES; c += blockDim.x) {
    float v1 = a1[c], v2 = a2[c];
    s1 += v1;
    s2 += v2;
    pp += (v1 - 1.0f) * (v2 - 1.0f);
  }

#pragma unroll
  for (int off = 16; off > 0; off >>= 1) {
    s1 += __shfl_down(s1, off, 32);
    s2 += __shfl_down(s2, off, 32);
    pp += __shfl_down(pp, off, 32);
  }

  __shared__ float ws1[8], ws2[8], wpp[8];
  __shared__ float bc[6];
  const int lane = threadIdx.x & 31;
  const int wid  = threadIdx.x >> 5;
  if (lane == 0) { ws1[wid] = s1; ws2[wid] = s2; wpp[wid] = pp; }
  __syncthreads();

  if (threadIdx.x == 0) {
    float S1 = 0.0f, S2 = 0.0f, P = 0.0f;
#pragma unroll
    for (int w = 0; w < 8; ++w) { S1 += ws1[w]; S2 += ws2[w]; P += wpp[w]; }
    const float K     = (float)CLASSES;
    const float u1    = K / S1;
    const float u2    = K / S2;
    const float sb1   = (S1 - K) / S1;
    const float sb2   = (S2 - K) / S2;
    const float diag  = P / (S1 * S2);
    const float conf  = sb1 * sb2 - diag;
    const float denom = 1.0f - conf;
    const float u_a   = (u1 * u2) / denom;
    const float S_a   = K / u_a;
    bc[0] = 1.0f / S1;
    bc[1] = 1.0f / S2;
    bc[2] = u1;
    bc[3] = u2;
    bc[4] = 1.0f / denom;
    bc[5] = S_a;
  }
  __syncthreads();

  const float iS1 = bc[0], iS2 = bc[1], u1 = bc[2], u2 = bc[3];
  const float idn = bc[4], Sa = bc[5];
  float* __restrict__ out_row = aa + (size_t)row * CLASSES;
  for (int c = threadIdx.x; c < CLASSES; c += blockDim.x) {
    float b1  = (a1[c] - 1.0f) * iS1;
    float b2  = (a2[c] - 1.0f) * iS2;
    float b_a = (b1 * b2 + b1 * u2 + b2 * u1) * idn;
    out_row[c] = b_a * Sa + 1.0f;
  }
}

// ===========================================================================
extern "C" void kernel_launch(void* const* d_in, const int* in_sizes, int n_in,
                              void* d_out, int out_size, void* d_ws, size_t ws_size,
                              hipStream_t stream) {
  (void)in_sizes; (void)n_in; (void)out_size;

  const float* x  = (const float*)d_in[0];
  const float* y  = (const float*)d_in[1];
  const float* Wx = (const float*)d_in[2];
  const float* bx = (const float*)d_in[3];
  const float* Wy = (const float*)d_in[4];
  const float* by = (const float*)d_in[5];

  float* out     = (float*)d_out;
  float* alpha_a = out;
  float* alpha_x = out + (size_t)BATCH * CLASSES;
  float* alpha_y = out + 2 * (size_t)BATCH * CLASSES;

  const size_t nX = (size_t)BATCH   * INPUT_DIM;   // 16,777,216 elems
  const size_t nW = (size_t)CLASSES * INPUT_DIM;   //  2,048,000 elems
  const size_t need = 2 * (nX + nW) * sizeof(__bf16);

  dim3 gemm_grid((CLASSES + 63) / 64, BATCH / 128, 2);  // (16, 64, 2)

  if (ws_size >= need && d_ws != nullptr) {
    __bf16* xb  = (__bf16*)d_ws;
    __bf16* yb  = xb  + nX;
    __bf16* Wxb = yb  + nX;
    __bf16* Wyb = Wxb + nW;

    const int nX8 = (int)(nX / 8), nW8 = (int)(nW / 8);
    cvt_f32_bf16_kernel<<<dim3((nX8 + 255) / 256), dim3(256), 0, stream>>>(x,  xb,  nX8);
    cvt_f32_bf16_kernel<<<dim3((nX8 + 255) / 256), dim3(256), 0, stream>>>(y,  yb,  nX8);
    cvt_f32_bf16_kernel<<<dim3((nW8 + 255) / 256), dim3(256), 0, stream>>>(Wx, Wxb, nW8);
    cvt_f32_bf16_kernel<<<dim3((nW8 + 255) / 256), dim3(256), 0, stream>>>(Wy, Wyb, nW8);

    gemm_softplus_bf16_kernel<<<gemm_grid, dim3(256), 0, stream>>>(
        xb, yb, Wxb, bx, Wyb, by, alpha_x, alpha_y);
  } else {
    gemm_softplus_fused_kernel<<<gemm_grid, dim3(256), 0, stream>>>(
        x, y, Wx, bx, Wy, by, alpha_x, alpha_y);
  }

  ds_combine_kernel<<<dim3(BATCH), dim3(256), 0, stream>>>(
      alpha_x, alpha_y, alpha_a);
}
